// TripleEncoder_44908178047265
// MI455X (gfx1250) — matrix operand order repaired and checked by reading, not compiled, not run on hardware
//
#include <hip/hip_runtime.h>
#include <hip/hip_bf16.h>

// ---------------- problem constants (from reference) ----------------
#define BB    16
#define MC    2048
#define MT    4096
#define EE    512
#define HOPS  2

typedef __attribute__((ext_vector_type(16))) __bf16 v16bf;
typedef __attribute__((ext_vector_type(8)))  float  v8f;
typedef __attribute__((ext_vector_type(4)))  __bf16 bf4;

// ---------------------------------------------------------------
// WMMA fragment assembly per CDNA5 16-bit VGPR layouts (wave32):
// A (16x32, MxK): lane<16 -> row M=lane,    elems[0..7]=kb+0..7,  [8..15]=kb+16..23
//                 lane>=16 -> row M=lane-16, elems[0..7]=kb+8..15, [8..15]=kb+24..31
// B (32x16, KxN): lane<16 -> col N=lane,    elems[0..7]=kb+0..7,  [8..15]=kb+16..23
//                 lane>=16 -> col N=lane-16, elems[0..7]=kb+8..15, [8..15]=kb+24..31
// B here is W^T: column n of B == row n of W (row-major, K contiguous)
// ---------------------------------------------------------------
__device__ __forceinline__ v16bf load_a16(const __bf16* __restrict__ base, int row, int kb, int lane) {
  int half = (lane >> 4) & 1;
  const __bf16* p = base + (size_t)row * EE + kb + half * 8;
  union { uint4 u; __bf16 b[8]; } lo, hi;
  lo.u = *(const uint4*)(p);
  hi.u = *(const uint4*)(p + 16);
  v16bf a;
#pragma unroll
  for (int i = 0; i < 8; ++i) { a[i] = lo.b[i]; a[i + 8] = hi.b[i]; }
  return a;
}

// B fragment from LDS-staged W slice: 64 rows x 256 K-cols, row stride 256
__device__ __forceinline__ v16bf lds_b16(const __bf16* s, int nlocal, int kbl, int lane) {
  int half = (lane >> 4) & 1;
  const __bf16* p = s + nlocal * 256 + kbl + half * 8;
  union { uint4 u; __bf16 b[8]; } lo, hi;
  lo.u = *(const uint4*)(p);
  hi.u = *(const uint4*)(p + 16);
  v16bf v;
#pragma unroll
  for (int i = 0; i < 8; ++i) { v[i] = lo.b[i]; v[i + 8] = hi.b[i]; }
  return v;
}

// async global -> LDS copy of one 16-byte chunk (GV mode, ASYNCcnt-tracked)
__device__ __forceinline__ void async_ld_b128(__bf16* lds_dst, const __bf16* gsrc) {
  unsigned           lds = (unsigned)(uintptr_t)lds_dst;       // LDS byte offset (addr[31:0])
  unsigned long long ga  = (unsigned long long)(uintptr_t)gsrc;
  asm volatile("global_load_async_to_lds_b128 %0, %1, off"
               :: "v"(lds), "v"(ga) : "memory");
}

#define WMMA_BF16(A, Bf, C) __builtin_amdgcn_wmma_f32_16x16x32_bf16( \
    false, (A), false, (Bf), (short)0, (C), false, false)

// ---------------------------------------------------------------
// GEMM: D[M x 512] = act( A1[M x 512] @ W1^T (+ A2 @ W2^T) )
// grid = (8 n-groups, M/256 m-blocks), 256 thr = 8 waves
// each wave: 32x64 output (2 M-tiles x 4 N-tiles), B staged in LDS
// staging uses global_load_async_to_lds_b128 + s_wait_asynccnt
// ---------------------------------------------------------------
__global__ __launch_bounds__(256) void te_gemm_bf16(
    const __bf16* __restrict__ A1, const __bf16* __restrict__ A2,
    const __bf16* __restrict__ W1, const __bf16* __restrict__ W2,
    float* __restrict__ D, int dorelu)
{
  __shared__ __align__(16) __bf16 sW1[64 * 256];   // 32 KB
  __shared__ __align__(16) __bf16 sW2[64 * 256];   // 32 KB

  const int wave = threadIdx.x >> 5;
  const int lane = threadIdx.x & 31;
  const int nb   = blockIdx.x << 6;                // N base (0..448)
  const int ncol = lane & 15;

  const int arow0 = (blockIdx.y << 8) + (wave << 5) + ncol;   // M-tile 0 row
  const int arow1 = arow0 + 16;                               // M-tile 1 row

  v8f acc0[4], acc1[4];
#pragma unroll
  for (int n = 0; n < 4; ++n)
#pragma unroll
    for (int i = 0; i < 8; ++i) { acc0[n][i] = 0.0f; acc1[n][i] = 0.0f; }

  for (int kh = 0; kh < 2; ++kh) {
    const int kc = kh << 8;                        // K chunk base (0, 256)
    __syncthreads();
    // async stage: W rows nb..nb+63, cols kc..kc+255 -> LDS (memory->LDS direct)
#pragma unroll
    for (int i = 0; i < 8; ++i) {
      int idx = threadIdx.x + (i << 8);
      int row = idx >> 5;
      int c   = (idx & 31) << 3;
      async_ld_b128(&sW1[row * 256 + c], W1 + (size_t)(nb + row) * EE + kc + c);
    }
    if (A2) {
#pragma unroll
      for (int i = 0; i < 8; ++i) {
        int idx = threadIdx.x + (i << 8);
        int row = idx >> 5;
        int c   = (idx & 31) << 3;
        async_ld_b128(&sW2[row * 256 + c], W2 + (size_t)(nb + row) * EE + kc + c);
      }
    }
    asm volatile("s_wait_asynccnt 0x0" ::: "memory");
    __syncthreads();

    for (int kt = 0; kt < 8; ++kt) {
      const int kbl = kt << 5;                     // K offset within chunk
      const int kb  = kc + kbl;                    // absolute K offset
      if (kt + 1 < 8)
        __builtin_prefetch(A1 + (size_t)arow0 * EE + kb + 32, 0, 0);

      v16bf a0 = load_a16(A1, arow0, kb, lane);
      v16bf a1 = load_a16(A1, arow1, kb, lane);
#pragma unroll
      for (int n = 0; n < 4; ++n) {
        v16bf bf = lds_b16(sW1, (n << 4) + ncol, kbl, lane);
        acc0[n] = WMMA_BF16(a0, bf, acc0[n]);
        acc1[n] = WMMA_BF16(a1, bf, acc1[n]);
      }
      if (A2) {
        v16bf c0 = load_a16(A2, arow0, kb, lane);
        v16bf c1 = load_a16(A2, arow1, kb, lane);
#pragma unroll
        for (int n = 0; n < 4; ++n) {
          v16bf bf = lds_b16(sW2, (n << 4) + ncol, kbl, lane);
          acc0[n] = WMMA_BF16(c0, bf, acc0[n]);
          acc1[n] = WMMA_BF16(c1, bf, acc1[n]);
        }
      }
    }
  }

  // C/D layout: VGPR j, lanes 0-15 -> M=j ; lanes 16-31 -> M=8+j ; N = lane%16
  const int rb0 = (blockIdx.y << 8) + (wave << 5) + ((lane >> 4) << 3);
#pragma unroll
  for (int n = 0; n < 4; ++n) {
    const int col = nb + (n << 4) + ncol;
#pragma unroll
    for (int j = 0; j < 8; ++j) {
      float v0 = acc0[n][j];
      float v1 = acc1[n][j];
      if (dorelu) { v0 = fmaxf(v0, 0.0f); v1 = fmaxf(v1, 0.0f); }
      D[(size_t)(rb0 + j) * EE + col]      = v0;
      D[(size_t)(rb0 + 16 + j) * EE + col] = v1;
    }
  }
}

// ---------------- message passing (scatter-add, f32 atomics) ----------------
__global__ __launch_bounds__(128) void te_messages(
    const float* __restrict__ h, const float* __restrict__ r,
    const int* __restrict__ head, const int* __restrict__ tail,
    const int* __restrict__ lab, float* __restrict__ upd, float* __restrict__ cnt)
{
  const int bt = blockIdx.x;              // b*MT + t
  if (lab[bt] == -1) return;
  const int b  = bt >> 12;                // / MT
  const int hi = head[bt];
  const int ti = tail[bt];
  const size_t hb = (size_t)b * MC;

  const float4* oh = (const float4*)(h + (hb + hi) * EE);
  const float4* ot = (const float4*)(h + (hb + ti) * EE);
  const float4* rr = (const float4*)(r + (size_t)bt * EE);
  float* ut = upd + (hb + ti) * EE;
  float* uh = upd + (hb + hi) * EE;

  const int e = threadIdx.x;              // 128 threads * float4 = 512
  float4 a = oh[e], c = ot[e], rv = rr[e];
  const int e4 = e << 2;
  __hip_atomic_fetch_add(&ut[e4 + 0], a.x - rv.x, __ATOMIC_RELAXED, __HIP_MEMORY_SCOPE_AGENT);
  __hip_atomic_fetch_add(&ut[e4 + 1], a.y - rv.y, __ATOMIC_RELAXED, __HIP_MEMORY_SCOPE_AGENT);
  __hip_atomic_fetch_add(&ut[e4 + 2], a.z - rv.z, __ATOMIC_RELAXED, __HIP_MEMORY_SCOPE_AGENT);
  __hip_atomic_fetch_add(&ut[e4 + 3], a.w - rv.w, __ATOMIC_RELAXED, __HIP_MEMORY_SCOPE_AGENT);
  __hip_atomic_fetch_add(&uh[e4 + 0], c.x - rv.x, __ATOMIC_RELAXED, __HIP_MEMORY_SCOPE_AGENT);
  __hip_atomic_fetch_add(&uh[e4 + 1], c.y - rv.y, __ATOMIC_RELAXED, __HIP_MEMORY_SCOPE_AGENT);
  __hip_atomic_fetch_add(&uh[e4 + 2], c.z - rv.z, __ATOMIC_RELAXED, __HIP_MEMORY_SCOPE_AGENT);
  __hip_atomic_fetch_add(&uh[e4 + 3], c.w - rv.w, __ATOMIC_RELAXED, __HIP_MEMORY_SCOPE_AGENT);
  if (e == 0) {
    __hip_atomic_fetch_add(&cnt[hb + ti], 1.0f, __ATOMIC_RELAXED, __HIP_MEMORY_SCOPE_AGENT);
    __hip_atomic_fetch_add(&cnt[hb + hi], 1.0f, __ATOMIC_RELAXED, __HIP_MEMORY_SCOPE_AGENT);
  }
}

// ---------------- small utility kernels ----------------
__global__ void te_zero4(float* __restrict__ p, int n4) {
  int i = blockIdx.x * blockDim.x + threadIdx.x;
  if (i < n4) ((float4*)p)[i] = make_float4(0.f, 0.f, 0.f, 0.f);
}

__global__ void te_cvt4(const float* __restrict__ s, __bf16* __restrict__ d, int n4) {
  int i = blockIdx.x * blockDim.x + threadIdx.x;
  if (i >= n4) return;
  float4 v = ((const float4*)s)[i];
  bf4 o;
  o[0] = (__bf16)v.x; o[1] = (__bf16)v.y; o[2] = (__bf16)v.z; o[3] = (__bf16)v.w;
  ((bf4*)d)[i] = o;
}

// A1 = bf16(h); A2 = bf16(upd / max(count,1))
__global__ void te_prep(const float* __restrict__ h, const float* __restrict__ upd,
                        const float* __restrict__ cnt,
                        __bf16* __restrict__ A1, __bf16* __restrict__ A2, int n4) {
  int i = blockIdx.x * blockDim.x + threadIdx.x;
  if (i >= n4) return;
  float4 hv = ((const float4*)h)[i];
  float4 uv = ((const float4*)upd)[i];
  float c = cnt[i >> 7];                  // (i*4)/512
  c = c < 1.0f ? 1.0f : c;
  bf4 o1, o2;
  o1[0] = (__bf16)hv.x; o1[1] = (__bf16)hv.y; o1[2] = (__bf16)hv.z; o1[3] = (__bf16)hv.w;
  o2[0] = (__bf16)(uv.x / c); o2[1] = (__bf16)(uv.y / c);
  o2[2] = (__bf16)(uv.z / c); o2[3] = (__bf16)(uv.w / c);
  ((bf4*)A1)[i] = o1;
  ((bf4*)A2)[i] = o2;
}

__global__ void te_gather(const float* __restrict__ tab, const int* __restrict__ ids,
                          float* __restrict__ dst, int n4) {
  int i = blockIdx.x * blockDim.x + threadIdx.x;
  if (i >= n4) return;
  int row = i >> 7;                       // / (EE/4)
  int c4  = i & 127;
  int id  = ids[row];
  ((float4*)dst)[i] = ((const float4*)(tab + (size_t)id * EE))[c4];
}

// out[b,t,:] = [ h[b,head] | r[b,t] | h[b,tail] ]
__global__ void te_concat(const float* __restrict__ h, const float* __restrict__ r,
                          const int* __restrict__ head, const int* __restrict__ tail,
                          float* __restrict__ out, int n4) {
  int i = blockIdx.x * blockDim.x + threadIdx.x;
  if (i >= n4) return;
  int row = i / 384;                      // 3*EE/4
  int c4  = i - row * 384;
  int seg = c4 >> 7;                      // /128
  int off = c4 & 127;
  int b   = row >> 12;                    // / MT
  float4 v;
  if (seg == 0)      v = ((const float4*)(h + ((size_t)b * MC + head[row]) * EE))[off];
  else if (seg == 1) v = ((const float4*)(r + (size_t)row * EE))[off];
  else               v = ((const float4*)(h + ((size_t)b * MC + tail[row]) * EE))[off];
  ((float4*)out)[i] = v;
}

// ---------------- host-side orchestration ----------------
extern "C" void kernel_launch(void* const* d_in, const int* in_sizes, int n_in,
                              void* d_out, int out_size, void* d_ws, size_t ws_size,
                              hipStream_t stream) {
  const float* concept_table  = (const float*)d_in[0];
  const float* relation_table = (const float*)d_in[1];
  const float* W_s            = (const float*)d_in[2];
  const float* W_n            = (const float*)d_in[3];
  const float* W_r            = (const float*)d_in[4];
  const int*   concept_ids    = (const int*)d_in[5];
  const int*   relation_ids   = (const int*)d_in[6];
  const int*   head_idx       = (const int*)d_in[7];
  const int*   tail_idx       = (const int*)d_in[8];
  const int*   labels         = (const int*)d_in[9];
  float* out = (float*)d_out;

  const size_t HN = (size_t)BB * MC * EE;     // 16,777,216
  const size_t RN = (size_t)BB * MT * EE;     // 33,554,432
  const size_t WN = (size_t)HOPS * EE * EE;   // 524,288 per weight tensor

  char* ws = (char*)d_ws;
  size_t off = 0;
  float*  h0   = (float*)(ws + off);  off += HN * 4;
  float*  h1   = (float*)(ws + off);  off += HN * 4;
  float*  upd  = (float*)(ws + off);  off += HN * 4;
  float*  r0   = (float*)(ws + off);  off += RN * 4;
  float*  r1   = (float*)(ws + off);  off += RN * 4;
  __bf16* A1   = (__bf16*)(ws + off); off += HN * 2;
  __bf16* A2   = (__bf16*)(ws + off); off += HN * 2;
  __bf16* rbf  = (__bf16*)(ws + off); off += RN * 2;
  __bf16* Wsbf = (__bf16*)(ws + off); off += WN * 2;
  __bf16* Wnbf = (__bf16*)(ws + off); off += WN * 2;
  __bf16* Wrbf = (__bf16*)(ws + off); off += WN * 2;
  float*  cnt  = (float*)(ws + off);  off += (size_t)BB * MC * 4;

  // weights -> bf16
  {
    int n4 = (int)(WN / 4);
    te_cvt4<<<(n4 + 255) / 256, 256, 0, stream>>>(W_s, Wsbf, n4);
    te_cvt4<<<(n4 + 255) / 256, 256, 0, stream>>>(W_n, Wnbf, n4);
    te_cvt4<<<(n4 + 255) / 256, 256, 0, stream>>>(W_r, Wrbf, n4);
  }

  // gathers
  {
    int n4 = (int)(HN / 4);
    te_gather<<<(n4 + 255) / 256, 256, 0, stream>>>(concept_table, concept_ids, h0, n4);
    n4 = (int)(RN / 4);
    te_gather<<<(n4 + 255) / 256, 256, 0, stream>>>(relation_table, relation_ids, r0, n4);
  }

  float* hin = h0; float* hout = h1;
  float* rin = r0; float* rout = r1;

  for (int l = 0; l < HOPS; ++l) {
    // zero upd + count
    {
      int n4 = (int)(HN / 4);
      te_zero4<<<(n4 + 255) / 256, 256, 0, stream>>>(upd, n4);
      int c4 = (BB * MC) / 4;
      te_zero4<<<(c4 + 255) / 256, 256, 0, stream>>>(cnt, c4);
    }
    // scatter-add messages
    te_messages<<<BB * MT, 128, 0, stream>>>(hin, rin, head_idx, tail_idx, labels, upd, cnt);
    // bf16 activation prep
    {
      int n4 = (int)(HN / 4);
      te_prep<<<(n4 + 255) / 256, 256, 0, stream>>>(hin, upd, cnt, A1, A2, n4);
      int r4 = (int)(RN / 4);
      te_cvt4<<<(r4 + 255) / 256, 256, 0, stream>>>(rin, rbf, r4);
    }
    // h_new = relu(A1 @ Ws^T + A2 @ Wn^T)   M = 32768 -> grid (8, 128)
    te_gemm_bf16<<<dim3(8, (BB * MC) / 256), 256, 0, stream>>>(
        A1, A2, Wsbf + (size_t)l * EE * EE, Wnbf + (size_t)l * EE * EE, hout, 1);
    // r_new = rbf @ Wr^T                    M = 65536 -> grid (8, 256)
    te_gemm_bf16<<<dim3(8, (BB * MT) / 256), 256, 0, stream>>>(
        rbf, (const __bf16*)nullptr, Wrbf + (size_t)l * EE * EE, (const __bf16*)nullptr,
        rout, 0);

    float* t;
    t = hin; hin = hout; hout = t;
    t = rin; rin = rout; rout = t;
  }

  // final concat: B x Mt x 3E
  {
    int n4 = (int)((size_t)BB * MT * 3 * EE / 4);
    te_concat<<<(n4 + 255) / 256, 256, 0, stream>>>(hin, rin, head_idx, tail_idx, out, n4);
  }
}
